// CausalSelfAttention_25572235281171
// MI455X (gfx1250) — matrix operand order repaired and checked by reading, not compiled, not run on hardware
//
#include <hip/hip_runtime.h>

// CDNA5 / gfx1250 causal self-attention: B=2, S=2048, E=1024, H=16, D=64.
// Pipeline: [QKV gemm -> f16 Q,K,V] -> [flash attention -> f16 Y] -> [proj gemm -> f32 out]
// Matmuls via v_wmma_f32_16x16x32_f16 (wave32); K tiles staged with
// global_load_async_to_lds_b128 (ASYNCcnt) in the attention kernel.

typedef __attribute__((ext_vector_type(16))) _Float16 v16h;
typedef __attribute__((ext_vector_type(8)))  _Float16 v8h;
typedef __attribute__((ext_vector_type(4)))  _Float16 v4h;
typedef __attribute__((ext_vector_type(8)))  float    v8f;
typedef __attribute__((ext_vector_type(4)))  float    v4f;

#define BN 2
#define SN 2048
#define EN 1024
#define HN 16
#define DN 64

__device__ __forceinline__ v8f wmma_f16(v16h a, v16h b, v8f c) {
  return __builtin_amdgcn_wmma_f32_16x16x32_f16(
      /*neg_a=*/false, a, /*neg_b=*/false, b,
      /*c_mod=*/(short)0, c, /*reuse_a=*/false, /*reuse_b=*/false);
}

// A-matrix fragment (16x32, f16). ISA layout: lane L (L<16) row M=L holds
// K = {0..7, 16..23}; lane L+16 holds row M=L, K = {8..15, 24..31}.
__device__ __forceinline__ v16h ldfrag_a(const _Float16* base, int ldh, int lane, int k0) {
  const int row = lane & 15;
  const int sel = lane >> 4;
  const _Float16* p = base + (size_t)row * ldh + k0 + 8 * sel;
  v8h lo = *(const v8h*)p;
  v8h hi = *(const v8h*)(p + 16);
  v16h r;
#pragma unroll
  for (int i = 0; i < 8; ++i) { r[i] = lo[i]; r[i + 8] = hi[i]; }
  return r;
}

// B-matrix fragment (32x16, f16) from a tile stored TRANSPOSED: tile[n][k].
// ISA layout: lane L holds column N=L&15; K = 8*(L>>4) + {0..7, 16..23}.
__device__ __forceinline__ v16h ldfrag_b(const _Float16* baseT, int ldh, int lane, int k0) {
  const int col = lane & 15;
  const int sel = lane >> 4;
  const _Float16* p = baseT + (size_t)col * ldh + k0 + 8 * sel;
  v8h lo = *(const v8h*)p;
  v8h hi = *(const v8h*)(p + 16);
  v16h r;
#pragma unroll
  for (int i = 0; i < 8; ++i) { r[i] = lo[i]; r[i + 8] = hi[i]; }
  return r;
}

__device__ __forceinline__ v4h cvt4(v4f f) {
  v4h h;
#pragma unroll
  for (int i = 0; i < 4; ++i) h[i] = (_Float16)f[i];
  return h;
}

// Async copy 16 bytes global -> LDS (tracked by ASYNCcnt).
__device__ __forceinline__ void async_g2l_b128(const _Float16* gaddr, _Float16* laddr) {
  const unsigned loff = (unsigned)(unsigned long long)laddr;  // low 32b = LDS byte addr
  asm volatile("global_load_async_to_lds_b128 %0, %1, off"
               :: "v"(loff), "v"((unsigned long long)gaddr)
               : "memory");
}
__device__ __forceinline__ void wait_asynccnt0() {
  asm volatile("s_wait_asynccnt 0x0" ::: "memory");
}

// ---------------------------------------------------------------------------
// Kernel 1: qkv = x @ W_attn + b_attn ; scatter into f16 Q,K,V [B,H,S,D].
// Block = 128 threads (4 waves). Tile: 128(M) x 64(N); wave owns 32x64.
// ---------------------------------------------------------------------------
__global__ void __launch_bounds__(128)
qkv_gemm_kernel(const float* __restrict__ x, const float* __restrict__ Wa,
                const float* __restrict__ ba,
                _Float16* __restrict__ Qb, _Float16* __restrict__ Kb,
                _Float16* __restrict__ Vb) {
  __shared__ alignas(16) _Float16 As[128][40];  // [m][k]
  __shared__ alignas(16) _Float16 Bs[64][40];   // transposed: [n][k]

  const int tid  = threadIdx.x;
  const int w    = tid >> 5;
  const int lane = tid & 31;
  const int n0   = blockIdx.x * 64;
  const int m0   = blockIdx.y * 128;

  v8f acc0[4] = {};
  v8f acc1[4] = {};

  for (int k0 = 0; k0 < EN; k0 += 32) {
    // ---- load phase (batched, no per-load waits) ----
    v4f fa[8];
#pragma unroll
    for (int it = 0; it < 8; ++it) {
      const int idx = tid + 128 * it;
      const int rr = idx >> 3, c4 = (idx & 7) * 4;
      fa[it] = *(const v4f*)(x + (size_t)(m0 + rr) * EN + k0 + c4);
    }
    v4f fb[4];
#pragma unroll
    for (int it = 0; it < 4; ++it) {
      const int idx = tid + 128 * it;
      const int kk = idx >> 4, n4 = (idx & 15) * 4;
      fb[it] = *(const v4f*)(Wa + (size_t)(k0 + kk) * (3 * EN) + n0 + n4);
    }
    if (k0 + 32 < EN) {
      __builtin_prefetch(x + (size_t)(m0 + tid) * EN + k0 + 32, 0, 1);
      __builtin_prefetch(Wa + (size_t)(k0 + 32 + (tid >> 2)) * (3 * EN) + n0 + (tid & 3) * 16, 0, 1);
    }
    // ---- convert + store phase ----
#pragma unroll
    for (int it = 0; it < 8; ++it) {
      const int idx = tid + 128 * it;
      const int rr = idx >> 3, c4 = (idx & 7) * 4;
      *(v4h*)&As[rr][c4] = cvt4(fa[it]);
    }
#pragma unroll
    for (int it = 0; it < 4; ++it) {
      const int idx = tid + 128 * it;
      const int kk = idx >> 4, n4 = (idx & 15) * 4;
      v4h h = cvt4(fb[it]);
#pragma unroll
      for (int t = 0; t < 4; ++t) Bs[n4 + t][kk] = h[t];
    }
    __syncthreads();

    const v16h a0 = ldfrag_a(&As[32 * w][0], 40, lane, 0);
    const v16h a1 = ldfrag_a(&As[32 * w + 16][0], 40, lane, 0);
#pragma unroll
    for (int j = 0; j < 4; ++j) {
      const v16h b = ldfrag_b(&Bs[16 * j][0], 40, lane, 0);
      acc0[j] = wmma_f16(a0, b, acc0[j]);
      acc1[j] = wmma_f16(a1, b, acc1[j]);
    }
    __syncthreads();
  }

  // Epilogue: bias + scatter. C layout: row = r + 8*(lane>>4), col = lane&15.
  const int hi = lane >> 4, ln = lane & 15;
#pragma unroll
  for (int g = 0; g < 2; ++g) {
#pragma unroll
    for (int j = 0; j < 4; ++j) {
#pragma unroll
      for (int r = 0; r < 8; ++r) {
        const int gm = m0 + 32 * w + 16 * g + r + 8 * hi;
        const int gn = n0 + 16 * j + ln;
        const float v = (g ? acc1[j][r] : acc0[j][r]) + ba[gn];
        const int bb = gm >> 11, ss = gm & (SN - 1);
        if (gn < EN) {
          const int hh = gn >> 6, dd = gn & 63;
          Qb[(((size_t)bb * HN + hh) * SN + ss) * DN + dd] = (_Float16)(v * 0.125f);
        } else if (gn < 2 * EN) {
          const int e = gn - EN, hh = e >> 6, dd = e & 63;
          Kb[(((size_t)bb * HN + hh) * SN + ss) * DN + dd] = (_Float16)v;
        } else {
          const int e = gn - 2 * EN, hh = e >> 6, dd = e & 63;
          Vb[(((size_t)bb * HN + hh) * SN + ss) * DN + dd] = (_Float16)v;
        }
      }
    }
  }
}

// ---------------------------------------------------------------------------
// Kernel 2: causal flash attention. grid = (S/64, B*H); block = 128 (4 waves).
// Wave w owns q rows [q0+16w, q0+16w+16). KV staged in 64-row blocks;
// K tile staged via async global->LDS copies (ASYNCcnt).
// ---------------------------------------------------------------------------
__global__ void __launch_bounds__(128)
attn_kernel(const _Float16* __restrict__ Qb, const _Float16* __restrict__ Kb,
            const _Float16* __restrict__ Vb, _Float16* __restrict__ Yb) {
  __shared__ alignas(16) _Float16 Ks[64][64];      // [kv][d]  (== B^T for Q@K^T)
  __shared__ alignas(16) _Float16 Vt[64][72];      // [d][kv]  (== B^T for P@V)
  __shared__ alignas(16) _Float16 Ps[4][16][72];   // per-wave P tile [q][kv]

  const int tid  = threadIdx.x;
  const int w    = tid >> 5;
  const int lane = tid & 31;
  const int hi   = lane >> 4, ln = lane & 15;
  const int q0   = blockIdx.x * 64;
  const int bh   = blockIdx.y;            // b*H + h
  const int b    = bh >> 4, h = bh & 15;

  const _Float16* Qp = Qb + (size_t)bh * SN * DN;
  const _Float16* Kp = Kb + (size_t)bh * SN * DN;
  const _Float16* Vp = Vb + (size_t)bh * SN * DN;

  // Q fragments for this wave (D=64 -> two k-steps), direct from global.
  const v16h aq0 = ldfrag_a(Qp + (size_t)(q0 + 16 * w) * DN, DN, lane, 0);
  const v16h aq1 = ldfrag_a(Qp + (size_t)(q0 + 16 * w) * DN, DN, lane, 32);

  v8f oacc[4] = {};
  float mstate[8], lstate[8];
#pragma unroll
  for (int r = 0; r < 8; ++r) { mstate[r] = -1.0e30f; lstate[r] = 0.0f; }

  const int jbmax = q0 >> 6;
  for (int jb = 0; jb <= jbmax; ++jb) {
    const _Float16* Kt = Kp + (size_t)jb * 64 * DN;
    const _Float16* Vv = Vp + (size_t)jb * 64 * DN;

    // K tile: async global -> LDS, f16 row-major [kv][d] (no conversion needed).
#pragma unroll
    for (int it = 0; it < 4; ++it) {
      const int i = tid + 128 * it;
      const int row = i >> 3, seg = (i & 7) * 8;
      async_g2l_b128(Kt + (size_t)row * DN + seg, &Ks[row][seg]);
    }
    // V tile: load phase, then register-transpose into Vt[d][kv].
    v8h vv[4];
#pragma unroll
    for (int it = 0; it < 4; ++it) {
      const int i = tid + 128 * it;
      const int row = i >> 3, seg = (i & 7) * 8;
      vv[it] = *(const v8h*)(Vv + (size_t)row * DN + seg);
    }
    if (jb < jbmax) {
      __builtin_prefetch(Kt + 64 * DN + tid * 32, 0, 1);
      __builtin_prefetch(Vv + 64 * DN + tid * 32, 0, 1);
    }
#pragma unroll
    for (int it = 0; it < 4; ++it) {
      const int i = tid + 128 * it;
      const int row = i >> 3, seg = (i & 7) * 8;
#pragma unroll
      for (int t = 0; t < 8; ++t) Vt[seg + t][row] = vv[it][t];
    }
    wait_asynccnt0();
    __syncthreads();

    // S = Q @ K^T  (16q x 64kv, K-dim = D = 64)
    v8f sacc[4] = {};
#pragma unroll
    for (int j = 0; j < 4; ++j) {
      v16h bk0 = ldfrag_b(&Ks[16 * j][0], 64, lane, 0);
      sacc[j] = wmma_f16(aq0, bk0, sacc[j]);
      v16h bk1 = ldfrag_b(&Ks[16 * j][0], 64, lane, 32);
      sacc[j] = wmma_f16(aq1, bk1, sacc[j]);
    }

    // Causal mask + online softmax (per q row r; 16-lane row groups).
#pragma unroll
    for (int r = 0; r < 8; ++r) {
      const int gq = q0 + 16 * w + r + 8 * hi;
      float mx = -1.0e30f;
#pragma unroll
      for (int j = 0; j < 4; ++j) {
        const int gk = jb * 64 + 16 * j + ln;
        float s = sacc[j][r];
        s = (gk <= gq) ? s : -1.0e30f;
        sacc[j][r] = s;
        mx = fmaxf(mx, s);
      }
#pragma unroll
      for (int t = 1; t < 16; t <<= 1) mx = fmaxf(mx, __shfl_xor(mx, t, 32));
      const float mnew = fmaxf(mstate[r], mx);
      const float corr = __expf(mstate[r] - mnew);
      mstate[r] = mnew;
      lstate[r] *= corr;
      float rs = 0.0f;
#pragma unroll
      for (int j = 0; j < 4; ++j) {
        const float p = __expf(sacc[j][r] - mnew);
        sacc[j][r] = p;
        rs += p;
      }
#pragma unroll
      for (int t = 1; t < 16; t <<= 1) rs += __shfl_xor(rs, t, 32);
      lstate[r] += rs;
#pragma unroll
      for (int j = 0; j < 4; ++j) oacc[j][r] *= corr;
    }

    // Re-shape P: C layout -> row-major LDS -> A fragments (wave-private, DS in-order).
#pragma unroll
    for (int j = 0; j < 4; ++j)
#pragma unroll
      for (int r = 0; r < 8; ++r)
        Ps[w][r + 8 * hi][16 * j + ln] = (_Float16)sacc[j][r];

    const v16h ap0 = ldfrag_a(&Ps[w][0][0], 72, lane, 0);
    const v16h ap1 = ldfrag_a(&Ps[w][0][0], 72, lane, 32);

    // O += P @ V  (16q x 64d, K-dim = 64kv)
#pragma unroll
    for (int j = 0; j < 4; ++j) {
      v16h bv0 = ldfrag_b(&Vt[16 * j][0], 72, lane, 0);
      oacc[j] = wmma_f16(ap0, bv0, oacc[j]);
      v16h bv1 = ldfrag_b(&Vt[16 * j][0], 72, lane, 32);
      oacc[j] = wmma_f16(ap1, bv1, oacc[j]);
    }
    __syncthreads();
  }

  // Normalize and write Y [B,S,E] f16: col = h*64 + d.
#pragma unroll
  for (int r = 0; r < 8; ++r) {
    const float inv = 1.0f / lstate[r];
    const int gq = q0 + 16 * w + r + 8 * hi;
#pragma unroll
    for (int j = 0; j < 4; ++j) {
      const int col = h * DN + 16 * j + ln;
      Yb[((size_t)b * SN + gq) * EN + col] = (_Float16)(oacc[j][r] * inv);
    }
  }
}

// ---------------------------------------------------------------------------
// Kernel 3: out = Y @ W_proj + b_proj  (f16 A from global, f32 out).
// Tile: 128(M) x 64(N); wave owns 32x64.
// ---------------------------------------------------------------------------
__global__ void __launch_bounds__(128)
proj_gemm_kernel(const _Float16* __restrict__ Yb, const float* __restrict__ Wp,
                 const float* __restrict__ bp, float* __restrict__ out) {
  __shared__ alignas(16) _Float16 Bs[64][40];   // transposed: [n][k]

  const int tid  = threadIdx.x;
  const int w    = tid >> 5;
  const int lane = tid & 31;
  const int n0   = blockIdx.x * 64;
  const int m0   = blockIdx.y * 128;

  v8f acc0[4] = {};
  v8f acc1[4] = {};

  for (int k0 = 0; k0 < EN; k0 += 32) {
    v4f fb[4];
#pragma unroll
    for (int it = 0; it < 4; ++it) {
      const int idx = tid + 128 * it;
      const int kk = idx >> 4, n4 = (idx & 15) * 4;
      fb[it] = *(const v4f*)(Wp + (size_t)(k0 + kk) * EN + n0 + n4);
    }
    if (k0 + 32 < EN) {
      __builtin_prefetch(Wp + (size_t)(k0 + 32 + (tid >> 2)) * EN + n0 + (tid & 3) * 16, 0, 1);
    }
#pragma unroll
    for (int it = 0; it < 4; ++it) {
      const int idx = tid + 128 * it;
      const int kk = idx >> 4, n4 = (idx & 15) * 4;
      v4h hh = cvt4(fb[it]);
#pragma unroll
      for (int t = 0; t < 4; ++t) Bs[n4 + t][kk] = hh[t];
    }
    __syncthreads();

    const v16h a0 = ldfrag_a(Yb + (size_t)(m0 + 32 * w) * EN + k0, EN, lane, 0);
    const v16h a1 = ldfrag_a(Yb + (size_t)(m0 + 32 * w + 16) * EN + k0, EN, lane, 0);
#pragma unroll
    for (int j = 0; j < 4; ++j) {
      const v16h b = ldfrag_b(&Bs[16 * j][0], 40, lane, 0);
      acc0[j] = wmma_f16(a0, b, acc0[j]);
      acc1[j] = wmma_f16(a1, b, acc1[j]);
    }
    __syncthreads();
  }

  const int hi = lane >> 4, ln = lane & 15;
#pragma unroll
  for (int g = 0; g < 2; ++g) {
#pragma unroll
    for (int j = 0; j < 4; ++j) {
#pragma unroll
      for (int r = 0; r < 8; ++r) {
        const int gm = m0 + 32 * w + 16 * g + r + 8 * hi;
        const int gn = n0 + 16 * j + ln;
        out[(size_t)gm * EN + gn] = (g ? acc1[j][r] : acc0[j][r]) + bp[gn];
      }
    }
  }
}

// ---------------------------------------------------------------------------
extern "C" void kernel_launch(void* const* d_in, const int* in_sizes, int n_in,
                              void* d_out, int out_size, void* d_ws, size_t ws_size,
                              hipStream_t stream) {
  const float* x  = (const float*)d_in[0];
  const float* Wa = (const float*)d_in[1];
  const float* ba = (const float*)d_in[2];
  const float* Wp = (const float*)d_in[3];
  const float* bp = (const float*)d_in[4];
  float* out = (float*)d_out;

  const size_t nQ = (size_t)BN * HN * SN * DN;   // 4M halves each
  _Float16* ws = (_Float16*)d_ws;
  _Float16* Qb = ws;
  _Float16* Kb = ws + nQ;
  _Float16* Vb = ws + 2 * nQ;
  _Float16* Yb = ws + 3 * nQ;

  qkv_gemm_kernel<<<dim3((3 * EN) / 64, (BN * SN) / 128), 128, 0, stream>>>(
      x, Wa, ba, Qb, Kb, Vb);
  attn_kernel<<<dim3(SN / 64, BN * HN), 128, 0, stream>>>(Qb, Kb, Vb, Yb);
  proj_gemm_kernel<<<dim3(EN / 64, (BN * SN) / 128), 128, 0, stream>>>(
      Yb, Wp, bp, out);
}
